// RNNLayer_188978561429
// MI455X (gfx1250) — compile-verified, matrix operand order checked by reading, not run on hardware
//
#include <hip/hip_runtime.h>
#include <hip/hip_bf16.h>

// ---------------------------------------------------------------------------
// RNN layer for MI455X (gfx1250, wave32, WMMA).
//   B=64, S=512, I=H=1024.
//   Phase 1: xi = x @ W_ih^T + b_ih  -> written time-major into d_out.
//   Phase 2: 512 sequential steps h = tanh(xi_t + h @ W_hh^T + b_hh).
//   Output layout (reference reshape trick): out[(t*B+b)*H + h].
//   h_final appended at offset B*S*H (contiguous copy of t=511 rows).
// Precision: f16 operands, f32 accumulate via v_wmma_f32_16x16x32_f16.
// Register tiling: 2x2 WMMA tiles/wave in phase 1 (1 load per WMMA),
//                  1x2 in phase 2 (keeps 16 blocks of parallelism).
// ---------------------------------------------------------------------------

typedef __attribute__((ext_vector_type(16))) _Float16 v16h;
typedef __attribute__((ext_vector_type(8)))  _Float16 v8h;
typedef __attribute__((ext_vector_type(8)))  float    v8f;
typedef __attribute__((ext_vector_type(4)))  float    v4f;

#define BATCH 64
#define SEQ   512
#define HID   1024   // == I == H == K

#define WMMA_F16(a, b, c) \
  __builtin_amdgcn_wmma_f32_16x16x32_f16(false, (a), false, (b), (short)0, (c), false, false)

// --- fragment loaders -------------------------------------------------------
// A-matrix 16x32 f16 layout (ISA 7.12.2):
//   lanes 0-15 : row M=lane,    halves = K 0..7  and 16..23
//   lanes 16-31: row M=lane-16, halves = K 8..15 and 24..31
// B-matrix (32x16) mirrors this with lanes <-> column N, halves <-> K, so the
// per-lane data for column n is W[n][k..] -- contiguous in row-major W.

__device__ inline v16h load_frag_f16(const _Float16* __restrict__ base,
                                     int ld, int lane) {
  const int r  = lane & 15;
  const int kk = (lane >> 4) * 8;
  const _Float16* p = base + (size_t)r * ld + kk;
  v8h lo = *(const v8h*)(p);        // K = kk .. kk+7      (16B)
  v8h hi = *(const v8h*)(p + 16);   // K = kk+16 .. kk+23  (16B)
  v16h out;
  ((v8h*)&out)[0] = lo;
  ((v8h*)&out)[1] = hi;
  return out;
}

__device__ inline v16h load_frag_f32(const float* __restrict__ base,
                                     size_t ld, int lane) {
  const int r  = lane & 15;
  const int kk = (lane >> 4) * 8;
  const float* p = base + (size_t)r * ld + kk;
  const v4f* q = (const v4f*)p;
  v4f f0 = q[0];   // K kk+0..3
  v4f f1 = q[1];   // K kk+4..7
  v4f f2 = q[4];   // K kk+16..19
  v4f f3 = q[5];   // K kk+20..23
  v16h out;
#pragma unroll
  for (int i = 0; i < 4; ++i) {
    out[i]      = (_Float16)f0[i];
    out[i + 4]  = (_Float16)f1[i];
    out[i + 8]  = (_Float16)f2[i];
    out[i + 12] = (_Float16)f3[i];
  }
  return out;
}

// --- utility kernels --------------------------------------------------------

__global__ void cvt_f32_to_f16_kernel(const float* __restrict__ in,
                                      _Float16* __restrict__ out, int n) {
  int i = blockIdx.x * blockDim.x + threadIdx.x;
  if (i < n) out[i] = (_Float16)in[i];
}

__global__ void zero_f16_kernel(_Float16* __restrict__ p, int n) {
  int i = blockIdx.x * blockDim.x + threadIdx.x;
  if (i < n) p[i] = (_Float16)0.0f;
}

// --- Phase 1: xi = x @ W_ih^T + b_ih, written time-major into out ----------
// Block: 256 threads = 8 waves (2x4), covering 64 output rows x 128 cols.
//   wave: wm = w>>2 (2 groups of 32 rows), wn = w&3 (4 groups of 32 cols).
// Each wave: 2x2 WMMA register tile (32x32), 4 accumulators.
// Output row r = t*64 + b (time-major); source x row = b*512 + t.
// A block's 64-row slab is exactly one timestep t -> fully regular addressing.

__global__ void gemm_xproj_kernel(const float* __restrict__ x,      // (B,S,I)
                                  const _Float16* __restrict__ Wih, // (H,K) f16
                                  const float* __restrict__ bih,    // (H)
                                  float* __restrict__ out) {
  const int lane = threadIdx.x & 31;
  const int wave = threadIdx.x >> 5;
  const int r0 = blockIdx.y * 64 + (wave >> 2) * 32;   // time-major row base
  const int n0 = blockIdx.x * 128 + (wave & 3) * 32;

  const int t  = r0 >> 6;        // timestep for this slab
  const int b0 = r0 & 63;        // batch base (0 or 32)
  const size_t lda = (size_t)SEQ * HID;                // x row stride (b-major)
  const float*    abase = x + ((size_t)b0 * SEQ + t) * HID;
  const _Float16* bbase = Wih + (size_t)n0 * HID;

  v8f c00 = {}, c01 = {}, c10 = {}, c11 = {};
#pragma unroll 2
  for (int k0 = 0; k0 < HID; k0 += 32) {
    v16h a0 = load_frag_f32(abase + k0, lda, lane);
    v16h a1 = load_frag_f32(abase + 16 * lda + k0, lda, lane);
    v16h w0 = load_frag_f16(bbase + k0, HID, lane);
    v16h w1 = load_frag_f16(bbase + (size_t)16 * HID + k0, HID, lane);
    c00 = WMMA_F16(a0, w0, c00);
    c01 = WMMA_F16(a0, w1, c01);
    c10 = WMMA_F16(a1, w0, c10);
    c11 = WMMA_F16(a1, w1, c11);
  }

  const int nloc = lane & 15;
  const int rloc = (lane >> 4) * 8;
  const float bv0 = bih[n0 + nloc];
  const float bv1 = bih[n0 + 16 + nloc];
#pragma unroll
  for (int j = 0; j < 8; ++j) {
    const size_t ra = (size_t)(r0 + rloc + j) * HID;
    const size_t rb = (size_t)(r0 + 16 + rloc + j) * HID;
    out[ra + n0 + nloc]      = c00[j] + bv0;
    out[ra + n0 + 16 + nloc] = c01[j] + bv1;
    out[rb + n0 + nloc]      = c10[j] + bv0;
    out[rb + n0 + 16 + nloc] = c11[j] + bv1;
  }
}

// --- Phase 2: one recurrent step -------------------------------------------
// Grid: 16 blocks x 256 threads. Block covers all 64 batch rows x 64 cols.
//   wave: m-tile = w&3 (4 tiles cover M=64), n-group = w>>2 (2 groups of 32).
// Each wave: 1x2 WMMA tile (16x32), A-fragment shared across both WMMAs.
// c = h_prev @ W_hh^T; out = tanh(out(=xi) + c + b_hh); h_next = (f16)out.

__global__ void rnn_step_kernel(const _Float16* __restrict__ hprev, // (B,H) f16
                                const _Float16* __restrict__ Whh,   // (H,H) f16
                                const float* __restrict__ bhh,      // (H)
                                float* __restrict__ out,
                                _Float16* __restrict__ hnext,       // (B,H) f16
                                int t) {
  const int lane = threadIdx.x & 31;
  const int wave = threadIdx.x >> 5;
  const int m0 = (wave & 3) * 16;
  const int n0 = blockIdx.x * 64 + (wave >> 2) * 32;

  const _Float16* abase = hprev + (size_t)m0 * HID;
  const _Float16* bbase = Whh + (size_t)n0 * HID;

  v8f c0 = {}, c1 = {};
#pragma unroll 2
  for (int k0 = 0; k0 < HID; k0 += 32) {
    v16h a  = load_frag_f16(abase + k0, HID, lane);
    v16h w0 = load_frag_f16(bbase + k0, HID, lane);
    v16h w1 = load_frag_f16(bbase + (size_t)16 * HID + k0, HID, lane);
    c0 = WMMA_F16(a, w0, c0);
    c1 = WMMA_F16(a, w1, c1);
  }

  const int nloc = lane & 15;
  const int brow = m0 + ((lane >> 4) * 8);   // batch row base for this lane
  const float bv0 = bhh[n0 + nloc];
  const float bv1 = bhh[n0 + 16 + nloc];
  const size_t orow = (size_t)t * BATCH + brow;
#pragma unroll
  for (int j = 0; j < 8; ++j) {
    const size_t o0 = (orow + j) * HID + n0 + nloc;
    const size_t o1 = (orow + j) * HID + n0 + 16 + nloc;
    const float v0 = tanhf(out[o0] + c0[j] + bv0);
    const float v1 = tanhf(out[o1] + c1[j] + bv1);
    out[o0] = v0;
    out[o1] = v1;
    hnext[(size_t)(brow + j) * HID + n0 + nloc]      = (_Float16)v0;
    hnext[(size_t)(brow + j) * HID + n0 + 16 + nloc] = (_Float16)v1;
  }
}

// --- launcher ---------------------------------------------------------------

extern "C" void kernel_launch(void* const* d_in, const int* in_sizes, int n_in,
                              void* d_out, int out_size, void* d_ws, size_t ws_size,
                              hipStream_t stream) {
  const float* x   = (const float*)d_in[0];  // (B,S,I)
  const float* Wih = (const float*)d_in[1];  // (H,I)
  const float* bih = (const float*)d_in[2];  // (H)
  const float* Whh = (const float*)d_in[3];  // (H,H)
  const float* bhh = (const float*)d_in[4];  // (H)
  float* out = (float*)d_out;

  // workspace: W_ih f16 (2MB) | W_hh f16 (2MB) | h0 (128KB) | h1 (128KB)
  _Float16* WihH  = (_Float16*)d_ws;
  _Float16* WhhH  = WihH + (size_t)HID * HID;
  _Float16* hbuf0 = WhhH + (size_t)HID * HID;
  _Float16* hbuf1 = hbuf0 + (size_t)BATCH * HID;

  const int HK = HID * HID;
  cvt_f32_to_f16_kernel<<<(HK + 255) / 256, 256, 0, stream>>>(Wih, WihH, HK);
  cvt_f32_to_f16_kernel<<<(HK + 255) / 256, 256, 0, stream>>>(Whh, WhhH, HK);
  zero_f16_kernel<<<(BATCH * HID + 255) / 256, 256, 0, stream>>>(hbuf0, BATCH * HID);

  // Phase 1: big GEMM, xi written time-major directly into d_out.
  dim3 g1(HID / 128, (BATCH * SEQ) / 64);  // 8 x 512 blocks
  gemm_xproj_kernel<<<g1, 256, 0, stream>>>(x, WihH, bih, out);

  // Phase 2: 512 sequential steps (stream order serializes them).
  _Float16* hb[2] = {hbuf0, hbuf1};
  for (int t = 0; t < SEQ; ++t) {
    rnn_step_kernel<<<HID / 64, 256, 0, stream>>>(hb[t & 1], WhhH, bhh, out,
                                                  hb[(t + 1) & 1], t);
  }

  // h_final: rows of t=511 are contiguous and already in (b,h) order.
  hipMemcpyAsync(out + (size_t)BATCH * SEQ * HID,
                 out + (size_t)(SEQ - 1) * BATCH * HID,
                 (size_t)BATCH * HID * sizeof(float),
                 hipMemcpyDeviceToDevice, stream);
}
